// PD_LSTM_38345468019409
// MI455X (gfx1250) — matrix operand order repaired and checked by reading, not compiled, not run on hardware
//
#include <hip/hip_runtime.h>

// ---------------------------------------------------------------------------
// Problem constants (match reference)
// ---------------------------------------------------------------------------
#define T_DIM 512
#define B_DIM 16
#define N_DIM 32
#define HD    256
#define NG    1024          // 4*HD
#define MAX_IT 20
#define BT    (T_DIM * B_DIM)   // 8192 rows

typedef __attribute__((ext_vector_type(16))) _Float16 v16h;
typedef __attribute__((ext_vector_type(8)))  float    v8f;

__device__ __forceinline__ float clip1(float v) {
    return fminf(1.0f, fmaxf(-1.0f, v));
}
__device__ __forceinline__ float sigmoidf_(float v) {
    return 1.0f / (1.0f + __expf(-v));
}

// A-fragment (16x32 f16) pack address for element (mm, kk) inside a tile:
//   lane = mm + 16*((kk>>3)&1),  j = (kk&7) + 8*((kk>>4)&1)
__device__ __forceinline__ int afrag_lane(int mm, int kk) {
    return mm + 16 * ((kk >> 3) & 1);
}
__device__ __forceinline__ int afrag_j(int kk) {
    return (kk & 7) + 8 * ((kk >> 4) & 1);
}

// ---------------------------------------------------------------------------
// Pack W (Ncols x K row-major; used as Y = X @ W^T) into WMMA B-fragment
// order: Bp[((kt*Ntiles + nt)*32 + lane)*16 + j]
// ---------------------------------------------------------------------------
__global__ __launch_bounds__(256) void pack_b_kernel(
    const float* __restrict__ W, _Float16* __restrict__ Bp, int Ncols, int K)
{
    int idx = blockIdx.x * 256 + threadIdx.x;
    if (idx >= Ncols * K) return;
    int col = idx / K, k = idx % K;
    int nt = col >> 4, n = col & 15;
    int kt = k >> 5,  kk = k & 31;
    int lane = n + 16 * ((kk >> 3) & 1);
    int j = (kk & 7) + 8 * ((kk >> 4) & 1);
    int Ntiles = Ncols >> 4;
    Bp[((kt * Ntiles + nt) * 32 + lane) * 16 + j] = (_Float16)W[idx];
}

// ---------------------------------------------------------------------------
// Pack X (M x K row-major f32) into WMMA A-fragment order:
//   Ap[((mt*ktiles + kt)*32 + lane)*16 + j]
// ---------------------------------------------------------------------------
__global__ __launch_bounds__(256) void pack_a_kernel(
    const float* __restrict__ X, _Float16* __restrict__ Ap, int M, int K)
{
    int idx = blockIdx.x * 256 + threadIdx.x;
    if (idx >= M * K) return;
    int m = idx / K, k = idx % K;
    int mt = m >> 4, mm = m & 15;
    int kt = k >> 5, kk = k & 31;
    int ktiles = K >> 5;
    Ap[((mt * ktiles + kt) * 32 + afrag_lane(mm, kk)) * 16 + afrag_j(kk)] =
        (_Float16)X[idx];
}

// ---------------------------------------------------------------------------
// Build M = I + tau*H (256x256, SPD) and invert via Gauss-Jordan.
// ---------------------------------------------------------------------------
__global__ __launch_bounds__(256) void build_invert_kernel(
    const float* __restrict__ H, const float* __restrict__ tau,
    float* __restrict__ Mw, float* __restrict__ Inv)
{
    __shared__ float rowM[256];
    __shared__ float rowI[256];
    int t = threadIdx.x;
    float tu = tau[0];
    for (int r = 0; r < 256; ++r) {
        Mw[r * 256 + t]  = ((r == t) ? 1.0f : 0.0f) + tu * H[r * 256 + t];
        Inv[r * 256 + t] = (r == t) ? 1.0f : 0.0f;
    }
    __syncthreads();
    for (int k = 0; k < 256; ++k) {
        float ip = 1.0f / Mw[k * 256 + k];
        float mk = Mw[k * 256 + t] * ip;
        float ik = Inv[k * 256 + t] * ip;
        rowM[t] = mk; rowI[t] = ik;
        Mw[k * 256 + t] = mk; Inv[k * 256 + t] = ik;
        __syncthreads();
        if (t != k) {
            float f = Mw[t * 256 + k];
            for (int c = 0; c < 256; ++c) {
                Mw[t * 256 + c]  -= f * rowM[c];
                Inv[t * 256 + c] -= f * rowI[c];
            }
        }
        __syncthreads();
    }
}

// ---------------------------------------------------------------------------
// WMMA GEMM with pre-packed f16 A and B fragments.
//   Y(MxN,f32) = Ap(MxK) @ Bp(KxN)    [+ bias0 + bias1]
// mode 0: Y = acc (+biases)
// mode 1: PD over-relaxation epilogue:
//         xc2 = acc; Y = xc2 + th*(xc2 - Xold); Xold = xc2
// One 16x16 output tile per wave; 8 waves / block.
// Inner loop per k-tile: 32B A load + 32B B load + one v_wmma.
// ---------------------------------------------------------------------------
__global__ __launch_bounds__(256) void wmma_gemm_kernel(
    const _Float16* __restrict__ Ap, const _Float16* __restrict__ Bp,
    int M, int K, int Ntiles,
    const float* __restrict__ bias0, const float* __restrict__ bias1,
    float* __restrict__ Y,
    const float* __restrict__ theta, float* __restrict__ Xold,
    int mode)
{
    int tile = blockIdx.x * 8 + (threadIdx.x >> 5);
    int lane = threadIdx.x & 31;
    int Mtiles = M >> 4;
    int mt = tile / Ntiles;
    int nt = tile - mt * Ntiles;
    if (mt >= Mtiles) return;                 // wave-uniform

    int ktiles = K >> 5;
    const _Float16* ap = &Ap[((mt * ktiles) * 32 + lane) * 16];
    const _Float16* bp = &Bp[(nt * 32 + lane) * 16];

    v8f acc = {};
    for (int kt = 0; kt < ktiles; ++kt) {
        v16h a = *(const v16h*)&ap[kt * (32 * 16)];
        v16h b = *(const v16h*)&bp[kt * (Ntiles * 32 * 16)];
        acc = __builtin_amdgcn_wmma_f32_16x16x32_f16(
            false, a, false, b, (short)0, acc, false, false);
    }

    int n  = lane & 15;
    int mh = (lane >> 4) * 8;
    int col = nt * 16 + n;
    int N = Ntiles * 16;

    if (mode == 0) {
        float bs = 0.0f;
        if (bias0) bs += bias0[col];
        if (bias1) bs += bias1[col];
#pragma unroll
        for (int v = 0; v < 8; ++v)
            Y[(mt * 16 + mh + v) * N + col] = acc[v] + bs;
    } else {
        float th = fminf(5.0f, fmaxf(0.0f, theta[0]));
#pragma unroll
        for (int v = 0; v < 8; ++v) {
            int idx = (mt * 16 + mh + v) * N + col;
            float xc2 = acc[v];
            float xo  = Xold[idx];
            Y[idx]    = xc2 + th * (xc2 - xo);
            Xold[idx] = xc2;
        }
    }
}

// ---------------------------------------------------------------------------
// Sequential LSTM scan (one persistent 32-wave workgroup).
// G   : (T*B, 1024) f32 precomputed input gates + biases, rows (t*16+b)
// Wp  : packed W_hh in B-fragment order, 8 k-tiles x 64 n-tiles
// Outputs (exactly one non-null):
//   hpack: h written as packed WMMA A-fragments, M-tile index = t
//          (feeds the next layer's input GEMM directly)
//   hf32 : h written f32 rows (b*T + t)  (the transposed "o" tensor)
// Dynamic LDS: gates 64KB | c 16KB | h packed A-frags 8KB  = 88KB
// ---------------------------------------------------------------------------
extern __shared__ char smem_raw[];

__global__ __launch_bounds__(1024) void lstm_scan_kernel(
    const float* __restrict__ G, const _Float16* __restrict__ Wp,
    _Float16* __restrict__ hpack, float* __restrict__ hf32)
{
    float*    ls_gates = (float*)smem_raw;                 // [16][1024]
    float*    ls_c     = (float*)(smem_raw + 65536);       // [16][256]
    _Float16* ls_h     = (_Float16*)(smem_raw + 81920);    // [8][32][16]

    const int tid  = threadIdx.x;
    const int wave = tid >> 5;
    const int lane = tid & 31;
    const int nt0  = wave * 2;                             // 2 n-tiles per wave

    for (int i = tid; i < 16 * 256;  i += 1024) ls_c[i] = 0.0f;
    for (int i = tid; i < 8 * 32 * 16; i += 1024) ls_h[i] = (_Float16)0.0f;

    // Warm L2/WGP$ with this wave's W_hh fragment region (global_prefetch_b8)
#pragma unroll
    for (int q = 0; q < 2; ++q)
#pragma unroll
        for (int kt = 0; kt < 8; ++kt)
            __builtin_prefetch(&Wp[((kt * 64 + nt0 + q) * 32 + lane) * 16], 0, 0);

    __syncthreads();

    for (int t = 0; t < T_DIM; ++t) {
        // ---- gate pre-activations: h @ W_hh^T via WMMA, + G[t] ----
        for (int q = 0; q < 2; ++q) {
            int nt = nt0 + q;
            v8f acc = {};
#pragma unroll
            for (int kt = 0; kt < 8; ++kt) {
                v16h a = *(const v16h*)&ls_h[(kt * 32 + lane) * 16];
                v16h b = *(const v16h*)&Wp[((kt * 64 + nt) * 32 + lane) * 16];
                acc = __builtin_amdgcn_wmma_f32_16x16x32_f16(
                    false, a, false, b, (short)0, acc, false, false);
            }
            int n  = lane & 15;
            int mh = (lane >> 4) * 8;
            int col = nt * 16 + n;
#pragma unroll
            for (int v = 0; v < 8; ++v) {
                int m = mh + v;
                ls_gates[m * NG + col] = acc[v] + G[(t * B_DIM + m) * NG + col];
            }
        }
        __syncthreads();

        // ---- fused i,f,g,o nonlinearity + state update (4096 elems) ----
#pragma unroll
        for (int p = 0; p < 4; ++p) {
            int idx = tid + p * 1024;
            int m = idx >> 8;
            int j = idx & 255;
            float gi = ls_gates[m * NG + j];
            float gf = ls_gates[m * NG + 256 + j];
            float gg = ls_gates[m * NG + 512 + j];
            float go = ls_gates[m * NG + 768 + j];
            float c  = sigmoidf_(gf) * ls_c[m * 256 + j]
                     + sigmoidf_(gi) * tanhf(gg);
            float h  = sigmoidf_(go) * tanhf(c);
            ls_c[m * 256 + j] = c;

            // packed fragment coordinates for h element (m, j)
            int kt = j >> 5, kk = j & 31;
            int pl = afrag_lane(m, kk);
            int pj = afrag_j(kk);
            _Float16 hh = (_Float16)h;
            ls_h[(kt * 32 + pl) * 16 + pj] = hh;

            if (hpack) {
                // rows are (t*16+m) => M-tile == t; 8 k-tiles of HD=256
                hpack[((t * 8 + kt) * 32 + pl) * 16 + pj] = hh;
            } else {
                hf32[(m * T_DIM + t) * HD + j] = h;
            }
        }
        __syncthreads();
    }
}

// ---------------------------------------------------------------------------
// Fused PD dual step + adjoint + rhs:
//   rhs = o - tau*adj(clip(x + sg*grad x)) + tau*bvec
// x layout (B,T,HD) flat; writes rhs directly as packed f16 A-fragments.
// ---------------------------------------------------------------------------
__global__ __launch_bounds__(256) void pd_rhs_kernel(
    const float* __restrict__ xt, const float* __restrict__ o,
    const float* __restrict__ bvec, const float* __restrict__ sigma,
    const float* __restrict__ tau, _Float16* __restrict__ rhsp)
{
    int idx = blockIdx.x * 256 + threadIdx.x;   // 0 .. B*T*HD-1
    int h = idx & 255;
    int s = (idx >> 8) & (T_DIM - 1);
    float sg = sigma[0], tu = tau[0];
    float xc = xt[idx];

    float y1c = 0.0f, y1p = 0.0f, y2c = 0.0f, y2p = 0.0f;
    if (s < T_DIM - 1) { float xn = xt[idx + HD]; y1c = clip1(xc + sg * (xn - xc)); }
    if (s > 0)         { float xq = xt[idx - HD]; y1p = clip1(xq + sg * (xc - xq)); }
    if (h < HD - 1)    { float xn = xt[idx + 1];  y2c = clip1(xc + sg * (xn - xc)); }
    if (h > 0)         { float xq = xt[idx - 1];  y2p = clip1(xq + sg * (xc - xq)); }

    float ladj = (y1p - y1c) + (y2p - y2c);
    float val = o[idx] - tu * ladj + tu * bvec[h];

    // pack into A-fragment layout: row = idx>>8, k = h; 8 k-tiles (HD=256)
    int row = idx >> 8;
    int mt = row >> 4, mm = row & 15;
    int kt = h >> 5,  kk = h & 31;
    rhsp[((mt * 8 + kt) * 32 + afrag_lane(mm, kk)) * 16 + afrag_j(kk)] =
        (_Float16)val;
}

// ---------------------------------------------------------------------------
// FC head on last timestep only: out = (x[:,T-1]@fc1^T+b1)@fc2^T+b2  (16x32)
// ---------------------------------------------------------------------------
__global__ __launch_bounds__(512) void head_kernel(
    const float* __restrict__ xt,
    const float* __restrict__ fc1w, const float* __restrict__ fc1b,
    const float* __restrict__ fc2w, const float* __restrict__ fc2b,
    float* __restrict__ out)
{
    __shared__ float tmp[16 * 256];
    int tid = threadIdx.x;
    for (int p = 0; p < 8; ++p) {
        int idx = tid + p * 512;
        int m = idx >> 8, j = idx & 255;
        const float* xr = &xt[(m * T_DIM + (T_DIM - 1)) * HD];
        float ssum = fc1b[j];
        for (int k = 0; k < 256; ++k) ssum += xr[k] * fc1w[j * 256 + k];
        tmp[m * 256 + j] = ssum;
    }
    __syncthreads();
    int m = tid >> 5, n = tid & 31;
    float ssum = fc2b[n];
    for (int k = 0; k < 256; ++k) ssum += tmp[m * 256 + k] * fc2w[n * 256 + k];
    out[m * 32 + n] = ssum;
}

// ---------------------------------------------------------------------------
// Orchestration
// ---------------------------------------------------------------------------
extern "C" void kernel_launch(void* const* d_in, const int* in_sizes, int n_in,
                              void* d_out, int out_size, void* d_ws, size_t ws_size,
                              hipStream_t stream)
{
    (void)in_sizes; (void)n_in; (void)out_size; (void)ws_size;

    const float* x     = (const float*)d_in[0];
    const float* w_ih0 = (const float*)d_in[2];
    const float* w_hh0 = (const float*)d_in[3];
    const float* b_ih0 = (const float*)d_in[4];
    const float* b_hh0 = (const float*)d_in[5];
    const float* w_ih1 = (const float*)d_in[6];
    const float* w_hh1 = (const float*)d_in[7];
    const float* b_ih1 = (const float*)d_in[8];
    const float* b_hh1 = (const float*)d_in[9];
    const float* fc1_w = (const float*)d_in[10];
    const float* fc1_b = (const float*)d_in[11];
    const float* fc2_w = (const float*)d_in[12];
    const float* fc2_b = (const float*)d_in[13];
    const float* Hmat  = (const float*)d_in[14];
    const float* bvec  = (const float*)d_in[15];
    const float* sigma = (const float*)d_in[16];
    const float* tau   = (const float*)d_in[17];
    const float* theta = (const float*)d_in[18];
    float* out = (float*)d_out;

    char* ws = (char*)d_ws;
    // workspace layout (bytes)
    float*    G      = (float*)(ws + 0);                    // 8192x1024 f32 (32MB, reused)
    _Float16* h1p    = (_Float16*)(ws + 33554432);          // 8192x256 f16 packed (4MB)
    float*    o_buf  = (float*)(ws + 41943040);             // 8192x256 f32 (B,T,HD)
    float*    x_t    = (float*)(ws + 50331648);
    float*    x_old  = (float*)(ws + 58720256);
    _Float16* rhsp   = (_Float16*)(ws + 67108864);          // 8192x256 f16 packed (4MB)
    float*    Mw     = (float*)(ws + 75497472);             // 256x256
    float*    AinvF  = (float*)(ws + 75759616);             // 256x256
    _Float16* p_wih0 = (_Float16*)(ws + 76021760);          // 1024x32  packed B
    _Float16* p_whh0 = (_Float16*)(ws + 76087296);          // 1024x256 packed B
    _Float16* p_wih1 = (_Float16*)(ws + 76611584);
    _Float16* p_whh1 = (_Float16*)(ws + 77135872);
    _Float16* p_ainv = (_Float16*)(ws + 77660160);          // 256x256 packed B
    _Float16* xp     = (_Float16*)(ws + 77791232);          // 8192x32 f16 packed A

    // ---- operand packing into WMMA fragment layouts ----
    pack_b_kernel<<<128,  256, 0, stream>>>(w_ih0, p_wih0, NG, N_DIM);
    pack_b_kernel<<<1024, 256, 0, stream>>>(w_hh0, p_whh0, NG, HD);
    pack_b_kernel<<<1024, 256, 0, stream>>>(w_ih1, p_wih1, NG, HD);
    pack_b_kernel<<<1024, 256, 0, stream>>>(w_hh1, p_whh1, NG, HD);
    pack_a_kernel<<<1024, 256, 0, stream>>>(x, xp, BT, N_DIM);

    // ---- (I + tau*H)^-1 once, then pack for the PD GEMM ----
    build_invert_kernel<<<1, 256, 0, stream>>>(Hmat, tau, Mw, AinvF);
    pack_b_kernel<<<256, 256, 0, stream>>>(AinvF, p_ainv, HD, HD);

    // ---- LSTM layer 0: batched input GEMM + sequential scan ----
    wmma_gemm_kernel<<<4096, 256, 0, stream>>>(
        xp, p_wih0, BT, N_DIM, 64, b_ih0, b_hh0, G, nullptr, nullptr, 0);
    lstm_scan_kernel<<<1, 1024, 90112, stream>>>(G, p_whh0, h1p, nullptr);

    // ---- LSTM layer 1 ----
    wmma_gemm_kernel<<<4096, 256, 0, stream>>>(
        h1p, p_wih1, BT, HD, 64, b_ih1, b_hh1, G, nullptr, nullptr, 0);
    lstm_scan_kernel<<<1, 1024, 90112, stream>>>(G, p_whh1, nullptr, o_buf);

    // ---- PD initialization: x_t = x_old = o ----
    hipMemcpyAsync(x_t,   o_buf, (size_t)BT * HD * 4, hipMemcpyDeviceToDevice, stream);
    hipMemcpyAsync(x_old, o_buf, (size_t)BT * HD * 4, hipMemcpyDeviceToDevice, stream);

    // ---- 20 primal-dual iterations ----
    for (int it = 0; it < MAX_IT; ++it) {
        pd_rhs_kernel<<<8192, 256, 0, stream>>>(x_t, o_buf, bvec, sigma, tau, rhsp);
        // xc2 = rhs @ Ainv^T ; x_t = xc2 + th*(xc2 - x_old); x_old = xc2
        wmma_gemm_kernel<<<1024, 256, 0, stream>>>(
            rhsp, p_ainv, BT, HD, 16, nullptr, nullptr, x_t, theta, x_old, 1);
    }

    // ---- FC head on last timestep ----
    head_kernel<<<1, 512, 0, stream>>>(x_t, fc1_w, fc1_b, fc2_w, fc2_b, out);
}
